// MultiHeadedAttentionProjection_54374285967840
// MI455X (gfx1250) — matrix operand order repaired and checked by reading, not compiled, not run on hardware
//
#include <hip/hip_runtime.h>

#define HH   8
#define DM   512
#define NB   128
#define SEQ  2048
#define NE   4096   // HH*DM
#define VSTR 40     // padded bf16 stride of transposed V tile (80B, 16B-aligned rows)

typedef __bf16 bf16_t;
typedef __attribute__((ext_vector_type(16))) __bf16    v16bf;
typedef __attribute__((ext_vector_type(8)))  __bf16    v8bf;
typedef __attribute__((ext_vector_type(8)))  float     v8f;
typedef __attribute__((ext_vector_type(8)))  unsigned  v8ui;

// ---- conversions ----
// two f32 -> packed bf16x2 in ONE v_perm_b32 (truncate): result = {hi[31:16], lo[31:16]}
__device__ __forceinline__ unsigned pack_bf2(float lo, float hi) {
    return __builtin_amdgcn_perm(__builtin_bit_cast(unsigned, hi),
                                 __builtin_bit_cast(unsigned, lo),
                                 0x07060302u);
}
// scalar f32 -> bf16 (RNE) for cold epilogue paths
__device__ __forceinline__ bf16_t f2bf(float f) {
    unsigned u = __builtin_bit_cast(unsigned, f);
    u = (u + 0x7FFFu + ((u >> 16) & 1u)) >> 16;
    unsigned short s = (unsigned short)u;
    return __builtin_bit_cast(bf16_t, s);
}

__device__ __forceinline__ v8f wmma_bf16(v16bf a, v16bf b, v8f c) {
    // 8 args: (neg_a, A, neg_b, B, c_mod, C, reuse_a, reuse_b)
    return __builtin_amdgcn_wmma_f32_16x16x32_bf16(false, a, false, b, (short)0, c,
                                                   false, false);
}

// A fragment (16x32 MxK) from bf16 row: base = row m + k0 + (lane>>4)*16
__device__ __forceinline__ v16bf load_a_bf16(const bf16_t* base) {
    v8bf lo = *(const v8bf*)(base);        // K = kb+0..7
    v8bf hi = *(const v8bf*)(base + 8);    // K = kb+8..15
    return __builtin_shufflevector(lo, hi, 0,1,2,3,4,5,6,7,8,9,10,11,12,13,14,15);
}

// A fragment from f32 row: base = row m + k0 + (lane>>4)*16  (8 v_perm)
__device__ __forceinline__ v16bf load_a_f32(const float* base) {
    const float4* p0 = (const float4*)(base);
    const float4* p1 = (const float4*)(base + 8);
    float4 x0 = p0[0], x1 = p0[1], x2 = p1[0], x3 = p1[1];
    v8ui u;
    u[0] = pack_bf2(x0.x, x0.y); u[1] = pack_bf2(x0.z, x0.w);
    u[2] = pack_bf2(x1.x, x1.y); u[3] = pack_bf2(x1.z, x1.w);
    u[4] = pack_bf2(x2.x, x2.y); u[5] = pack_bf2(x2.z, x2.w);
    u[6] = pack_bf2(x3.x, x3.y); u[7] = pack_bf2(x3.z, x3.w);
    return __builtin_bit_cast(v16bf, u);
}

// B fragment (32x16 KxN): lane holds column n = lane&15, 16 consecutive K at
// base = source row n + k0 + (lane>>4)*16  (4x b128 load + 8 v_perm)
__device__ __forceinline__ v16bf load_b_f32(const float* base) {
    const float4* p = (const float4*)base;
    float4 x0 = p[0], x1 = p[1], x2 = p[2], x3 = p[3];
    v8ui u;
    u[0] = pack_bf2(x0.x, x0.y); u[1] = pack_bf2(x0.z, x0.w);
    u[2] = pack_bf2(x1.x, x1.y); u[3] = pack_bf2(x1.z, x1.w);
    u[4] = pack_bf2(x2.x, x2.y); u[5] = pack_bf2(x2.z, x2.w);
    u[6] = pack_bf2(x3.x, x3.y); u[7] = pack_bf2(x3.z, x3.w);
    return __builtin_bit_cast(v16bf, u);
}

// ---------------- Kernel 1: q projection  q_ws(bf16) = query * W_in^T + b_in --------------
__global__ void qproj_kernel(const float* __restrict__ query,
                             const float* __restrict__ W_in,
                             const float* __restrict__ b_in,
                             bf16_t* __restrict__ q_ws) {
    int lane = threadIdx.x & 31, wave = threadIdx.x >> 5;
    int tid = blockIdx.x * 8 + wave;           // 0..2047
    int nt = tid >> 3, mt = tid & 7;
    int m0 = mt * 16, n0 = nt * 16;
    int nl = lane & 15, kg = lane >> 4;
    const float* arow = query + (size_t)(m0 + nl) * DM + kg * 16;
    const float* brow = W_in  + (size_t)(n0 + nl) * DM + kg * 16;
    v8f acc = {};
#pragma unroll 4
    for (int k0 = 0; k0 < DM; k0 += 32) {
        v16bf a = load_a_f32(arow + k0);
        v16bf b = load_b_f32(brow + k0);
        acc = wmma_bf16(a, b, acc);
    }
    int n = n0 + nl;
    float bias = b_in[n];
    int rbase = kg * 8;
#pragma unroll
    for (int r = 0; r < 8; ++r) {
        int row = m0 + rbase + r;
        q_ws[(size_t)row * NE + n] = f2bf(acc[r] + bias);
    }
}

// ---------------- Kernel 2: attention per batch ----------------
__global__ void attn_kernel(const bf16_t* __restrict__ q_ws,
                            const float* __restrict__ key,
                            const float* __restrict__ value,
                            const int* __restrict__ mask,
                            bf16_t* __restrict__ x_ws) {
    extern __shared__ char smem[];
    bf16_t* lds_q  = (bf16_t*)smem;                                   // 16*512  bf16 (16 KB)
    float*  lds_s  = (float*)(smem + 16*DM*2);                        // 8*2048  f32  (64 KB)
    bf16_t* lds_p  = (bf16_t*)(smem + 16*DM*2 + 8*SEQ*4);             // 16*2048 bf16 (64 KB)
    bf16_t* lds_vT = (bf16_t*)(smem + 16*DM*2 + 8*SEQ*4 + 16*SEQ*2);  // 512*VSTR bf16 (40 KB)

    int b = blockIdx.x;
    int lane = threadIdx.x & 31, wave = threadIdx.x >> 5;
    int nl = lane & 15, kg = lane >> 4;

    // stage Q (8 heads padded to 16 rows) as bf16 in LDS
    for (int i = threadIdx.x; i < 16 * DM; i += 256) {
        int row = i >> 9;
        lds_q[i] = (row < HH) ? q_ws[(size_t)b * NE + i] : f2bf(0.0f);
    }
    __syncthreads();

    // ---- Phase 1: scores = Q K^T / sqrt(d), masked ----
    const float scale = 0.04419417382415922f;             // 1/sqrt(512)
    const bf16_t* qp = lds_q + (size_t)nl * DM + kg * 16;
    for (int t = wave; t < SEQ / 16; t += 8) {
        int s0 = t * 16;
        const float* kp = key + ((size_t)b * SEQ + s0 + nl) * DM + kg * 16;
        v8f acc = {};
#pragma unroll 4
        for (int k0 = 0; k0 < DM; k0 += 32) {
            v16bf a  = load_a_bf16(qp + k0);
            v16bf bm = load_b_f32(kp + k0);
            acc = wmma_bf16(a, bm, acc);
        }
        if (kg == 0) {                                    // lanes 0..15 hold rows 0..7
            int s = s0 + nl;
            bool dead = (mask[(size_t)b * SEQ + s] == 0);
#pragma unroll
            for (int r = 0; r < 8; ++r)
                lds_s[r * SEQ + s] = dead ? -1e9f : acc[r] * scale;
        }
    }
    __syncthreads();

    // ---- Phase 2: softmax (one wave per head); p -> bf16 pairs, rows 8..15 zeroed ----
    {
        int h = wave;
        float mx = -3.0e38f;
        for (int s = lane; s < SEQ; s += 32) mx = fmaxf(mx, lds_s[h * SEQ + s]);
        for (int off = 16; off; off >>= 1) mx = fmaxf(mx, __shfl_xor(mx, off, 32));
        float sum = 0.f;
        for (int s = lane; s < SEQ; s += 32) {
            float e = __expf(lds_s[h * SEQ + s] - mx);
            lds_s[h * SEQ + s] = e;
            sum += e;
        }
        for (int off = 16; off; off >>= 1) sum += __shfl_xor(sum, off, 32);
        float inv = 1.0f / sum;
        unsigned* prow  = (unsigned*)lds_p + h * (SEQ / 2);
        unsigned* pzrow = (unsigned*)lds_p + (h + 8) * (SEQ / 2);
        for (int s2 = lane; s2 < SEQ / 2; s2 += 32) {
            float e0 = lds_s[h * SEQ + 2 * s2]     * inv;
            float e1 = lds_s[h * SEQ + 2 * s2 + 1] * inv;
            prow[s2]  = pack_bf2(e0, e1);
            pzrow[s2] = 0u;
        }
    }
    __syncthreads();

    // ---- Phase 3: X = P V. V chunk staged TRANSPOSED in LDS (b128 ds reads) ----
    v8f acc[4] = {{}, {}, {}, {}};                        // wave owns d-tiles wave+8j
    unsigned* vTw = (unsigned*)lds_vT;
    for (int s0 = 0; s0 < SEQ; s0 += 32) {
        // stage: idx = r2*512 + c ; thread reads value[s0+2*r2][c], value[s0+2*r2+1][c]
        for (int i = threadIdx.x; i < 16 * DM; i += 256) {
            int r2 = i >> 9, c = i & (DM - 1);
            const float* vp = value + ((size_t)b * SEQ + s0 + 2 * r2) * DM + c;
            vTw[c * (VSTR / 2) + r2] = pack_bf2(vp[0], vp[DM]);
        }
        __syncthreads();
        v16bf a = load_a_bf16(lds_p + (size_t)nl * SEQ + s0 + kg * 16);
#pragma unroll
        for (int j = 0; j < 4; ++j) {
            int col = (wave + 8 * j) * 16 + nl;           // output d index
            const bf16_t* vb = lds_vT + (size_t)col * VSTR + kg * 16;
            v8bf lo = *(const v8bf*)vb;
            v8bf hi = *(const v8bf*)(vb + 8);
            v16bf bm = __builtin_shufflevector(lo, hi, 0,1,2,3,4,5,6,7,
                                               8,9,10,11,12,13,14,15);
            acc[j] = wmma_bf16(a, bm, acc[j]);
        }
        __syncthreads();
    }
    if (kg == 0) {                                        // rows 0..7 = real heads
#pragma unroll
        for (int j = 0; j < 4; ++j) {
            int d = (wave + 8 * j) * 16 + nl;
#pragma unroll
            for (int r = 0; r < 8; ++r)
                x_ws[(size_t)b * NE + r * DM + d] = f2bf(acc[j][r]);
        }
    }
}

// ---------------- Kernel 3: output projection  out = x * W_out^T + b_out --------------
__global__ void outproj_kernel(const bf16_t* __restrict__ x_ws,
                               const float* __restrict__ W_out,
                               const float* __restrict__ b_out,
                               float* __restrict__ out) {
    int lane = threadIdx.x & 31, wave = threadIdx.x >> 5;
    int tid = blockIdx.x * 8 + wave;           // 0..2047
    int nt = tid >> 3, mt = tid & 7;
    int m0 = mt * 16, n0 = nt * 16;
    int nl = lane & 15, kg = lane >> 4;
    const bf16_t* arow = x_ws  + (size_t)(m0 + nl) * NE + kg * 16;
    const float*  brow = W_out + (size_t)(n0 + nl) * NE + kg * 16;
    v8f acc = {};
#pragma unroll 8
    for (int k0 = 0; k0 < NE; k0 += 32) {
        v16bf a = load_a_bf16(arow + k0);
        v16bf b = load_b_f32(brow + k0);
        acc = wmma_bf16(a, b, acc);
    }
    int n = n0 + nl;
    float bias = b_out[n];
    int rbase = kg * 8;
#pragma unroll
    for (int r = 0; r < 8; ++r) {
        int row = m0 + rbase + r;
        out[(size_t)row * NE + n] = acc[r] + bias;
    }
}

extern "C" void kernel_launch(void* const* d_in, const int* in_sizes, int n_in,
                              void* d_out, int out_size, void* d_ws, size_t ws_size,
                              hipStream_t stream) {
    const float* query = (const float*)d_in[0];
    const float* key   = (const float*)d_in[1];
    const float* value = (const float*)d_in[2];
    const int*   mask  = (const int*)d_in[3];
    const float* W_in  = (const float*)d_in[4];
    const float* b_in  = (const float*)d_in[5];
    const float* W_out = (const float*)d_in[6];
    const float* b_out = (const float*)d_in[7];
    float* out = (float*)d_out;

    bf16_t* q_ws = (bf16_t*)d_ws;                       // 128*4096 bf16 = 1 MB
    bf16_t* x_ws = q_ws + (size_t)NB * NE;              // 128*4096 bf16 = 1 MB

    // 2048 tiles / 8 waves per block -> 256 blocks of 256 threads
    qproj_kernel<<<256, 256, 0, stream>>>(query, W_in, b_in, q_ws);

    // one block per batch; LDS: Q(16K) + S(64K) + P(64K) + Vt(40K) = 184 KB of 320 KB WGP
    size_t attn_lds = (size_t)16 * DM * 2 + (size_t)8 * SEQ * 4 +
                      (size_t)16 * SEQ * 2 + (size_t)DM * VSTR * 2;
    attn_kernel<<<NB, 256, attn_lds, stream>>>(q_ws, key, value, mask, x_ws);

    outproj_kernel<<<256, 256, 0, stream>>>(x_ws, W_out, b_out, out);
}
